// GCN_24919400251509
// MI455X (gfx1250) — matrix operand order repaired and checked by reading, not compiled, not run on hardware
//
#include <hip/hip_runtime.h>
#include <hip/hip_bf16.h>

// ---------------------------------------------------------------------------
// CDNA5 (gfx1250) WMMA types
// ---------------------------------------------------------------------------
typedef __bf16        v16bf __attribute__((ext_vector_type(16)));
typedef float         v8f   __attribute__((ext_vector_type(8)));

static __device__ __forceinline__ unsigned int pack2bf(float a, float b) {
    union { __bf16 h[2]; unsigned int u; } p;
    p.h[0] = (__bf16)a; p.h[1] = (__bf16)b;
    return p.u;
}

// ---------------------------------------------------------------------------
// Dense GEMM  H[M,NC] = A[M,K] @ W[K,NC]   (fp32 in, bf16 WMMA, fp32 out)
//
// Block = 256 threads = 8 wave32s; each wave owns one 16x16 output tile.
//   WAVES_N = NC/16 column tiles, WAVES_M = 8/WAVES_N row tiles per block.
//
// LDS staging (fragment-friendly):
//   ldsA  : A row-tile, row-major [BROWS][K] bf16  -> A fragment = two
//           16B-aligned runs of 8 bf16  => 2x ds_load_b128 per K-step.
//   ldsWt : W TRANSPOSED, [NC][K] bf16 (K contiguous per column) -> B
//           fragment = 16 consecutive bf16 at one column => 2x ds_load_b128.
//
// 16-bit fragment layouts used here, contiguous k order (lane l, kg=l>>4, ln=l&15):
//   A: VGPR v holds pair k0 = ks + kg*16 + 2v, row = ln
//   B: VGPR v holds rows kk = ks + kg*16 + 2v, kk+1 at col = ln
//   C/D: VGPR v -> row (v + 8*kg), col = ln
//
// Row-range guard is uniform per block: only the single tail block takes the
// guarded path; all full blocks use unguarded loads/stores with immediate
// offsets off one base address.
// ---------------------------------------------------------------------------
template <int K, int NC>
__global__ __launch_bounds__(256)
void gemm_wmma(const float* __restrict__ A, const float* __restrict__ W,
               float* __restrict__ H, int M) {
    constexpr int WAVES_N = NC / 16;
    constexpr int WAVES_M = 8 / WAVES_N;
    constexpr int BROWS   = WAVES_M * 16;

    __shared__ __attribute__((aligned(16))) __bf16 ldsA[BROWS * K];
    __shared__ __attribute__((aligned(16))) __bf16 ldsWt[NC * K];

    const int  t        = threadIdx.x;
    const int  rowBase  = blockIdx.x * BROWS;
    const bool fullTile = (rowBase + BROWS) <= M;

    // --- stage A row-tile: float4 global loads -> packed bf16x4 ds_store_b64
    constexpr int AQUADS = (BROWS * K) / (256 * 4);   // 4
    uint2* ldsA2 = (uint2*)ldsA;
    if (fullTile) {
        const float4* Ab = (const float4*)(A + (long long)rowBase * K);
#pragma unroll
        for (int i = 0; i < AQUADS; ++i) {
            int q = t + i * 256;
            float4 v = Ab[q];
            uint2 p;
            p.x = pack2bf(v.x, v.y);
            p.y = pack2bf(v.z, v.w);
            ldsA2[q] = p;
        }
    } else {
#pragma unroll
        for (int i = 0; i < AQUADS; ++i) {
            int q  = t + i * 256;
            int el = q * 4;
            int r  = el / K, c = el % K;
            int gr = rowBase + r;
            float4 v = make_float4(0.f, 0.f, 0.f, 0.f);
            if (gr < M) v = *(const float4*)(A + (long long)gr * K + c);
            uint2 p;
            p.x = pack2bf(v.x, v.y);
            p.y = pack2bf(v.z, v.w);
            ldsA2[q] = p;
        }
    }

    // --- stage W transposed: 4 strided K reads per thread -> one ds_store_b64
    constexpr int KQ     = K / 4;
    constexpr int WQUADS = (K * NC) / (256 * 4);      // 8 or 2
    uint2* ldsWt2 = (uint2*)ldsWt;
#pragma unroll
    for (int i = 0; i < WQUADS; ++i) {
        int g  = t + i * 256;           // (col, k-quad) group
        int c  = g / KQ;
        int kq = g % KQ;
        const float* wp = W + (long long)(4 * kq) * NC + c;
        float w0 = wp[0 * NC], w1 = wp[1 * NC], w2 = wp[2 * NC], w3 = wp[3 * NC];
        uint2 p;
        p.x = pack2bf(w0, w1);
        p.y = pack2bf(w2, w3);
        ldsWt2[g] = p;                  // ldsWt[c*K + 4*kq .. +3]
    }
    __syncthreads();

    const int wave = t >> 5;
    const int lane = t & 31;
    const int wm   = wave / WAVES_N;
    const int wn   = wave % WAVES_N;
    const int ln   = lane & 15;
    const int kg   = lane >> 4;

    const int arow = wm * 16 + ln;      // A fragment row for this lane
    const int bcol = wn * 16 + ln;      // B fragment col for this lane

    const uint4* a16 = (const uint4*)ldsA;
    const uint4* w16 = (const uint4*)ldsWt;

    v8f acc = {};
#pragma unroll
    for (int ks = 0; ks < K; ks += 32) {
        union { uint4 q[2]; v16bf b; } ua, ub;
        // A fragment: two 8-bf16 runs (16B aligned)
        int abase = arow * K + ks + kg * 16;
        ua.q[0] = a16[(abase) >> 3];
        ua.q[1] = a16[(abase + 8) >> 3];
        // B fragment: 16 consecutive bf16 at column bcol (32B)
        int bbase = bcol * K + ks + kg * 16;
        ub.q[0] = w16[(bbase) >> 3];
        ub.q[1] = w16[(bbase + 8) >> 3];

        acc = __builtin_amdgcn_wmma_f32_16x16x32_bf16(
            /*neg_a=*/false, ua.b, /*neg_b=*/false, ub.b,
            /*c_mod=*/(short)0, acc, /*reuse_a=*/false, /*reuse_b=*/false);
    }

    // --- store 16x16 f32 tile: one base pointer, immediate row offsets ---
    const long long rbase = (long long)rowBase + wm * 16 + kg * 8;
    float* p = H + rbase * NC + wn * 16 + ln;
    if (fullTile) {
#pragma unroll
        for (int v = 0; v < 8; ++v) p[v * NC] = acc[v];
    } else {
#pragma unroll
        for (int v = 0; v < 8; ++v)
            if (rbase + v < M) p[v * NC] = acc[v];
    }
}

// ---------------------------------------------------------------------------
// agg[i*NC + j] = b[j]  (bias broadcast; float4-vectorized, NC multiple of 4)
// ---------------------------------------------------------------------------
__global__ void bias_init4(float4* __restrict__ out, const float4* __restrict__ b,
                           int total4, int NC4m1) {
    int i = blockIdx.x * blockDim.x + threadIdx.x;
    if (i < total4) out[i] = b[i & NC4m1];   // NC/4 is a power of two
}

__global__ void relu_ip4(float4* __restrict__ x, int n4) {
    int i = blockIdx.x * blockDim.x + threadIdx.x;
    if (i < n4) {
        float4 v = x[i];
        v.x = fmaxf(v.x, 0.f); v.y = fmaxf(v.y, 0.f);
        v.z = fmaxf(v.z, 0.f); v.w = fmaxf(v.w, 0.f);
        x[i] = v;
    }
}

// ---------------------------------------------------------------------------
// Edge scatter:  agg[dst] += h[src] * w     (NC/4 lanes per edge; float4
// gather from L2-resident h, per-element global_atomic_add_f32 into agg)
// ---------------------------------------------------------------------------
template <int NC>
__global__ __launch_bounds__(256)
void scatter_edges(const float* __restrict__ h, const int* __restrict__ src,
                   const int* __restrict__ dst, const float* __restrict__ w,
                   float* __restrict__ agg, int nEdges) {
    constexpr int LPE = NC / 4;
    int tid = blockIdx.x * blockDim.x + threadIdx.x;
    int e   = tid / LPE;
    int sub = tid % LPE;
    if (e >= nEdges) return;

    int   s  = src[e];
    int   d  = dst[e];
    float wt = w[e];

    const float4* h4 = (const float4*)h;
    float4 v = h4[(long long)s * LPE + sub];

    float* base = agg + (long long)d * NC + sub * 4;
    atomicAdd(base + 0, v.x * wt);
    atomicAdd(base + 1, v.y * wt);
    atomicAdd(base + 2, v.z * wt);
    atomicAdd(base + 3, v.w * wt);
}

// ---------------------------------------------------------------------------
// Launch: gemm1 -> init(b1) -> scatter1 -> relu -> gemm2 -> init(b2) -> scatter2
// ---------------------------------------------------------------------------
extern "C" void kernel_launch(void* const* d_in, const int* in_sizes, int n_in,
                              void* d_out, int out_size, void* d_ws, size_t ws_size,
                              hipStream_t stream) {
    const float* x    = (const float*)d_in[0];
    const int*   esrc = (const int*)d_in[1];
    const int*   edst = (const int*)d_in[2];
    const float* ew   = (const float*)d_in[3];
    const float* W1   = (const float*)d_in[4];
    const float* b1   = (const float*)d_in[5];
    const float* W2   = (const float*)d_in[6];
    const float* b2   = (const float*)d_in[7];
    float*       out  = (float*)d_out;

    const int hidden = in_sizes[5];                 // 64
    const int m      = in_sizes[4] / hidden;        // 128
    const int N      = in_sizes[0] / m;             // 100000
    const int E      = in_sizes[1];                 // 1600000
    (void)n_in; (void)out_size; (void)ws_size;

    float* h1   = (float*)d_ws;                     // N*64
    float* agg1 = h1 + (size_t)N * hidden;          // N*64
    float* h2   = h1;                               // reuse (h1 dead after scatter1)

    // ---- layer 1: h1 = x @ W1 ----
    gemm_wmma<128, 64><<<(N + 31) / 32, 256, 0, stream>>>(x, W1, h1, N);

    // ---- agg1 = b1 (broadcast) ----
    {
        int total4 = N * 64 / 4;
        bias_init4<<<(total4 + 255) / 256, 256, 0, stream>>>(
            (float4*)agg1, (const float4*)b1, total4, 16 - 1);
    }
    // ---- agg1[dst] += h1[src] * w ----
    {
        long long threads = (long long)E * 16;
        scatter_edges<64><<<(int)((threads + 255) / 256), 256, 0, stream>>>(
            h1, esrc, edst, ew, agg1, E);
    }
    // ---- relu in place ----
    {
        int n4 = N * 64 / 4;
        relu_ip4<<<(n4 + 255) / 256, 256, 0, stream>>>((float4*)agg1, n4);
    }

    // ---- layer 2: h2 = agg1 @ W2 ----
    gemm_wmma<64, 32><<<(N + 63) / 64, 256, 0, stream>>>(agg1, W2, h2, N);

    // ---- out = b2 (broadcast) ----
    {
        int total4 = N * 32 / 4;
        bias_init4<<<(total4 + 255) / 256, 256, 0, stream>>>(
            (float4*)out, (const float4*)b2, total4, 8 - 1);
    }
    // ---- out[dst] += h2[src] * w ----
    {
        long long threads = (long long)E * 8;
        scatter_edges<32><<<(int)((threads + 255) / 256), 256, 0, stream>>>(
            h2, esrc, edst, ew, out, E);
    }
}
